// LinearAttention_10608569221172
// MI455X (gfx1250) — matrix operand order repaired and checked by reading, not compiled, not run on hardware
//
#include <hip/hip_runtime.h>

// ---------------------------------------------------------------------------
// MI455X (gfx1250) implementation, 4 kernels:
//   1) pack W f32 -> bf16 in WMMA-B lane layout            (ws +0,   6 MB)
//   2) pack x f32 -> bf16 row-major                        (ws +6MB, 32 MB)
//   3) tiled GEMM qkv = x@W + b, 256x256 tiles, bf16 WMMA,
//      A K-slabs staged to LDS via TDM tensor_load_to_lds  (ws +38MB, 192 MB f32)
//   4) per-token attention epilogue (softmax over head dim,
//      attn = (q*s)K^T, out = attn V) with f32 WMMA 16x16x4
// ---------------------------------------------------------------------------

typedef __bf16 bf16;
typedef bf16  v16bf __attribute__((ext_vector_type(16)));
typedef float v8f   __attribute__((ext_vector_type(8)));
typedef float v2f   __attribute__((ext_vector_type(2)));
typedef unsigned int v4u __attribute__((ext_vector_type(4)));
typedef int   v8i   __attribute__((ext_vector_type(8)));
typedef int   v4i   __attribute__((ext_vector_type(4)));

#define DIM    1024
#define DIM3   3072
#define SCALE  0.125f      // HEAD_DIM^-0.5, HEAD_DIM = 64
#define USE_TDM 1

// ---- workspace layout (bytes) ----
#define WP_OFF   0ull                        // 6 MB  : packed W bf16
#define XBF_OFF  (6ull << 20)                // 32 MB : x bf16 row-major [tokens][1024]
#define QKV_OFF  (XBF_OFF + (32ull << 20))   // 192 MB: qkv f32 [tokens][3072]

__device__ __forceinline__ unsigned bf_bits(float f) {
  unsigned u = __float_as_uint(f);
  return (u + 0x7FFFu + ((u >> 16) & 1u)) >> 16;   // f32 -> bf16 RNE
}

struct U32x8 { uint4 lo, hi; };

__device__ __forceinline__ v16bf wmma_bf16_frag(const unsigned short* p) {
  U32x8 r = { *(const uint4*)p, *(const uint4*)(p + 8) };
  return __builtin_bit_cast(v16bf, r);
}

// ---------------------------------------------------------------------------
// Kernel 1: W[1024][3072] f32 -> wp packed bf16:
//   wp[(((tn*32+ks)*32+lane)*16 + j)] = bf16( W[ks*32 + (lane>>4)*16 + j][tn*16 + (lane&15)] )
// so each lane's 16 B elements for one (tile, k-step) are 32 contiguous bytes.
// ---------------------------------------------------------------------------
__global__ __launch_bounds__(256) void pack_w_bf16(const float* __restrict__ W,
                                                   unsigned short* __restrict__ wp) {
  int flat = blockIdx.x * 256 + threadIdx.x;
  int j  = flat & 15;
  int l  = (flat >> 4) & 31;
  int ks = (flat >> 9) & 31;
  int tn = flat >> 14;
  int n  = tn * 16 + (l & 15);
  int k  = ks * 32 + (l >> 4) * 16 + j;
  wp[flat] = (unsigned short)bf_bits(W[(size_t)k * DIM3 + n]);
}

// Kernel 2: x f32 -> bf16, flat copy (each thread converts 4 elements)
__global__ __launch_bounds__(256) void pack_x_bf16(const float* __restrict__ x,
                                                   unsigned short* __restrict__ xb) {
  size_t i = ((size_t)blockIdx.x * 256 + threadIdx.x) * 4;
  float4 v = *(const float4*)(x + i);
  *(uint2*)(xb + i) = make_uint2(bf_bits(v.x) | (bf_bits(v.y) << 16),
                                 bf_bits(v.z) | (bf_bits(v.w) << 16));
}

// ---------------------------------------------------------------------------
// TDM descriptor: move one 256-row x 64-byte K-slab of bf16 A into LDS.
// D# per CDNA5 ISA: data_size=8B, 2D tile 8 x 256,
// tensor_dim0=256 (8B units, row pitch 2048B), tensor_dim0_stride=256.
// ---------------------------------------------------------------------------
#if USE_TDM
__device__ __forceinline__ void tdm_load_xslab(const unsigned short* rowbase,
                                               int ks, unsigned lds_off) {
  unsigned long long ga = (unsigned long long)(uintptr_t)rowbase
                        + (unsigned long long)ks * 64ull;
  v4u g0;
  g0.x = 1u;                                             // count=1 (valid user D#)
  g0.y = lds_off;                                        // LDS byte address
  g0.z = (unsigned)ga;                                   // global_addr[31:0]
  g0.w = ((unsigned)(ga >> 32) & 0x1FFFFFFu) | 0x80000000u; // addr[56:32] | type=2
  v8i g1;
  g1[0] = 0x00030000;   // workgroup_mask=0, data_size=3 (8 bytes)
  g1[1] = 0x01000000;   // tensor_dim0 = 256  (bits 63:48)
  g1[2] = 0x01000000;   // tensor_dim1 = 256  (bits 95:80)
  g1[3] = 0x00080000;   // tile_dim0   = 8    (bits 127:112)
  g1[4] = 0x00000100;   // tile_dim1   = 256
  g1[5] = 0x00000100;   // tensor_dim0_stride = 256 (8B units = 2048B row pitch)
  g1[6] = 0;
  g1[7] = 0;
  v4i z4 = {0, 0, 0, 0};
  v8i z8 = {0, 0, 0, 0, 0, 0, 0, 0};
  // 6-arg form (clang-23 / therock-10.0 headers): (g0, g1, g2, g3, g4, cpol)
  __builtin_amdgcn_tensor_load_to_lds(g0, g1, z4, z4, z8, 0);
}
#endif

// ---------------------------------------------------------------------------
// Kernel 3: GEMM qkv = x @ W + bias.  Grid (tokens/256, 3072/256), 512 threads.
// Wave (wm,wn) in 4x4 grid owns a 64x64 sub-tile: 4x4 frags of v8f.
// A from LDS double buffer (TDM staged), B register-direct from packed wp (L2).
// All 4 B frags get distinct registers so their 8 b128 loads issue in one
// clause and the waits stagger (loadcnt 3->0), hiding L2 latency under WMMA.
// ---------------------------------------------------------------------------
__global__ __launch_bounds__(512) void gemm_qkv(const unsigned short* __restrict__ xb,
                                                const unsigned short* __restrict__ wp,
                                                const float* __restrict__ bias,
                                                float* __restrict__ qkv) {
  __shared__ unsigned short Abuf[2][256 * 32];   // 2 x 16 KB K-slabs

  const int tid  = threadIdx.x;
  const int lane = tid & 31;
  const int w    = tid >> 5;
  const int half = lane >> 4;
  const int col  = lane & 15;
  const int wm   = w & 3;
  const int wn   = w >> 2;
  const int bx   = blockIdx.x, by = blockIdx.y;

  const unsigned short* xrow0 = xb + (size_t)bx * 256 * DIM;
  const unsigned short* wpB   = wp + (size_t)(by * 16 + wn * 4) * 16384 + lane * 16;

  v8f acc[4][4] = {};

#if USE_TDM
  if (w == 0) {
    tdm_load_xslab(xrow0, 0, (unsigned)(uintptr_t)&Abuf[0][0]);
    __builtin_amdgcn_s_wait_tensorcnt(0);
  }
  __syncthreads();
#else
  {
    const unsigned short* src = xrow0 + (size_t)(tid >> 1) * DIM + (tid & 1) * 16;
    *(uint4*)&Abuf[0][(tid >> 1) * 32 + (tid & 1) * 16]     = *(const uint4*)src;
    *(uint4*)&Abuf[0][(tid >> 1) * 32 + (tid & 1) * 16 + 8] = *(const uint4*)(src + 8);
  }
  __syncthreads();
#endif

  for (int ks = 0; ks < 32; ++ks) {
    const int cur = ks & 1;
#if USE_TDM
    if (w == 0 && ks + 1 < 32)
      tdm_load_xslab(xrow0, ks + 1, (unsigned)(uintptr_t)&Abuf[cur ^ 1][0]);
#endif
    // Issue all B loads first (8 x global_load_b128, distinct registers)
    const unsigned short* bp = wpB + ks * 512;
    v16bf Bf[4];
    #pragma unroll
    for (int ni = 0; ni < 4; ++ni)
      Bf[ni] = wmma_bf16_frag(bp + (size_t)ni * 16384);

    // A frags (16x32 bf16): lane half=0 -> K 0-15 ; half=1 -> K 16-31
    v16bf Af[4];
    #pragma unroll
    for (int mi = 0; mi < 4; ++mi)
      Af[mi] = wmma_bf16_frag(&Abuf[cur][(wm * 64 + mi * 16 + col) * 32 + half * 16]);

    #pragma unroll
    for (int ni = 0; ni < 4; ++ni)
      #pragma unroll
      for (int mi = 0; mi < 4; ++mi)
        acc[mi][ni] = __builtin_amdgcn_wmma_f32_16x16x32_bf16(
            false, Af[mi], false, Bf[ni], (short)0, acc[mi][ni], false, false);

#if USE_TDM
    if (w == 0 && ks + 1 < 32) __builtin_amdgcn_s_wait_tensorcnt(0);
    __syncthreads();
#else
    __syncthreads();
    if (ks + 1 < 32) {
      const unsigned short* src =
          xrow0 + (size_t)(tid >> 1) * DIM + (ks + 1) * 32 + (tid & 1) * 16;
      *(uint4*)&Abuf[cur ^ 1][(tid >> 1) * 32 + (tid & 1) * 16]     = *(const uint4*)src;
      *(uint4*)&Abuf[cur ^ 1][(tid >> 1) * 32 + (tid & 1) * 16 + 8] = *(const uint4*)(src + 8);
    }
    __syncthreads();
#endif
  }

  // bias + store (D layout: row = r + 8*half, col = lane&15)
  #pragma unroll
  for (int ni = 0; ni < 4; ++ni) {
    const int nc = (by * 16 + wn * 4 + ni) * 16 + col;
    const float bb = bias[nc];
    #pragma unroll
    for (int mi = 0; mi < 4; ++mi) {
      const int row0 = bx * 256 + wm * 64 + mi * 16 + 8 * half;
      #pragma unroll
      for (int r = 0; r < 8; ++r)
        qkv[(size_t)(row0 + r) * DIM3 + nc] = acc[mi][ni][r] + bb;
    }
  }
}

// ---------------------------------------------------------------------------
// Kernel 4: per-token attention. One WG per 16 tokens; slab -> LDS; then
// softmax over head dim of k, attn = (q*s)Ksm^T, out = attn V (f32 WMMA 16x16x4).
// ---------------------------------------------------------------------------
#define ATTN_SMEM (16 * DIM3 * 4 + 8 * 256 * 4)   // 192 KB slab + 8 KB attn scratch
static_assert(ATTN_SMEM <= 320 * 1024, "exceeds WGP LDS");

__global__ __launch_bounds__(256) void attn_ep(const float* __restrict__ qkvg,
                                               float* __restrict__ out) {
  extern __shared__ char smem[];
  float* qkv    = (float*)smem;
  float* attn_s = (float*)(smem + 16 * DIM3 * 4);

  const int tid  = threadIdx.x;
  const int lane = tid & 31;
  const int w    = tid >> 5;
  const int half = lane >> 4;
  const int col  = lane & 15;

  // stage 16-token qkv slab into LDS
  const float* src = qkvg + (size_t)blockIdx.x * 16 * DIM3;
  #pragma unroll
  for (int i = 0; i < 48; ++i)
    *(float4*)(qkv + i * 1024 + tid * 4) = *(const float4*)(src + i * 1024 + tid * 4);
  __syncthreads();

  for (int tt = 0; tt < 2; ++tt) {
    const int t = w * 2 + tt;
    float* tok = qkv + t * DIM3;

    // softmax over head dim of k: lane -> (head = lane>>1, 32-elem half)
    {
      const int h = lane >> 1, part = lane & 1;
      float* krow = tok + DIM + h * 64 + part * 32;
      float vals[32];
      float mx = -3.4e38f;
      #pragma unroll
      for (int i = 0; i < 8; ++i) {
        float4 f = *(const float4*)(krow + i * 4);
        vals[4*i+0] = f.x; vals[4*i+1] = f.y; vals[4*i+2] = f.z; vals[4*i+3] = f.w;
        mx = fmaxf(mx, fmaxf(fmaxf(f.x, f.y), fmaxf(f.z, f.w)));
      }
      mx = fmaxf(mx, __shfl_xor(mx, 1, 32));
      float s = 0.f;
      #pragma unroll
      for (int i = 0; i < 32; ++i) { vals[i] = __expf(vals[i] - mx); s += vals[i]; }
      s += __shfl_xor(s, 1, 32);
      const float inv = 1.0f / s;
      #pragma unroll
      for (int i = 0; i < 8; ++i)
        *(float4*)(krow + i * 4) = make_float4(vals[4*i+0]*inv, vals[4*i+1]*inv,
                                               vals[4*i+2]*inv, vals[4*i+3]*inv);
    }
    __syncthreads();   // uniform: every wave runs this loop exactly twice

    // attn = Q(16x64) * Ksm^T(64x16) via 16 x wmma f32 16x16x4
    v8f aacc = {};
    const float* qb = tok + col * 64;
    const float* kb = tok + DIM + col * 64;
    #pragma unroll
    for (int kk = 0; kk < 16; ++kk) {
      const int o = kk * 4 + half * 2;
      v2f a = *(const v2f*)(qb + o);
      v2f b = *(const v2f*)(kb + o);
      aacc = __builtin_amdgcn_wmma_f32_16x16x4_f32(
          false, a, false, b, (short)0, aacc, false, false);
    }
    float* as = attn_s + w * 256;
    #pragma unroll
    for (int r = 0; r < 8; ++r)
      as[(r + 8 * half) * 16 + col] = aacc[r] * SCALE;
    __syncthreads();

    // out = attn(16x16) * V(16x64): 4 N-tiles x 4 K-steps
    v8f oacc[4] = {};
    #pragma unroll
    for (int kk = 0; kk < 4; ++kk) {
      const int o = kk * 4 + half * 2;
      v2f a = *(const v2f*)(as + col * 16 + o);
      #pragma unroll
      for (int nt = 0; nt < 4; ++nt) {
        v2f b;
        b[0] = tok[2048 + (o + 0) * 64 + nt * 16 + col];
        b[1] = tok[2048 + (o + 1) * 64 + nt * 16 + col];
        oacc[nt] = __builtin_amdgcn_wmma_f32_16x16x4_f32(
            false, a, false, b, (short)0, oacc[nt], false, false);
      }
    }
    const size_t tg = (size_t)blockIdx.x * 16 + t;
    #pragma unroll
    for (int nt = 0; nt < 4; ++nt)
      #pragma unroll
      for (int r = 0; r < 8; ++r)
        out[tg * DIM + (r + 8 * half) * 64 + nt * 16 + col] = oacc[nt][r];
  }
}

// ---------------------------------------------------------------------------
extern "C" void kernel_launch(void* const* d_in, const int* in_sizes, int n_in,
                              void* d_out, int out_size, void* d_ws, size_t ws_size,
                              hipStream_t stream) {
  (void)n_in; (void)out_size; (void)ws_size;   // needs ws_size >= 230 MB
  const float* x  = (const float*)d_in[0];
  const float* Wq = (const float*)d_in[1];
  const float* bq = (const float*)d_in[2];
  float* out = (float*)d_out;

  unsigned short* wp  = (unsigned short*)((char*)d_ws + WP_OFF);
  unsigned short* xbf = (unsigned short*)((char*)d_ws + XBF_OFF);
  float*          qkv = (float*)((char*)d_ws + QKV_OFF);

  const int tokens = in_sizes[0] / DIM;            // B*N = 16384

  pack_w_bf16<<<(DIM * DIM3) / 256, 256, 0, stream>>>(Wq, wp);
  pack_x_bf16<<<(tokens * DIM) / 1024, 256, 0, stream>>>(x, xbf);

  dim3 ggrid(tokens / 256, DIM3 / 256);
  gemm_qkv<<<ggrid, 512, 0, stream>>>(xbf, wp, bq, qkv);

  (void)hipFuncSetAttribute((const void*)attn_ep,
                            hipFuncAttributeMaxDynamicSharedMemorySize,
                            (int)ATTN_SMEM);
  attn_ep<<<tokens / 16, 256, ATTN_SMEM, stream>>>(qkv, out);
}
